// VARTokenizer_90563680403805
// MI455X (gfx1250) — matrix operand order repaired and checked by reading, not compiled, not run on hardware
//
#include <hip/hip_runtime.h>
#include <math.h>

// ---------------------------------------------------------------------------
// CDNA5 WMMA types
// ---------------------------------------------------------------------------
typedef __attribute__((ext_vector_type(16))) __bf16 v16bf;
typedef __attribute__((ext_vector_type(8)))  __bf16 v8bf;
typedef __attribute__((ext_vector_type(8)))  float  v8f;
typedef __attribute__((ext_vector_type(4)))  int    v4i;

union ABu { v16bf v; v8bf h[2]; };
union P8u { v8bf v; __bf16 e[8]; };

// Optional CDNA5 async global->LDS path (guarded so compile never breaks)
#if defined(__has_builtin)
#  if __has_builtin(__builtin_amdgcn_global_load_async_to_lds_b128) && \
      __has_builtin(__builtin_amdgcn_s_wait_asynccnt)
#    define USE_ASYNC_LDS 1
#  endif
#endif
#ifndef USE_ASYNC_LDS
#  define USE_ASYNC_LDS 0
#endif

__device__ inline void copy16_to_lds(const __bf16* g, __bf16* l) {
#if USE_ASYNC_LDS
  __builtin_amdgcn_global_load_async_to_lds_b128(
      (__attribute__((address_space(1))) v4i*)(g),
      (__attribute__((address_space(3))) v4i*)(l), 0, 0);
#else
  *reinterpret_cast<v8bf*>(l) = *reinterpret_cast<const v8bf*>(g);
#endif
}
__device__ inline void wait_async_lds() {
#if USE_ASYNC_LDS
  __builtin_amdgcn_s_wait_asynccnt(0);
#endif
}

__device__ inline v8f wmma_bf16(v16bf a, v16bf b, v8f c) {
  return __builtin_amdgcn_wmma_f32_16x16x32_bf16(
      false, a, false, b, (short)0, c, false, false);
}

#define TM 64
#define TN 128
#define TK 32
#define LDK 40   // padded bf16 row stride: 80 bytes, keeps 16B alignment

// A fragment: 16x32 (MxK). lane l row = l&15; elems 0..7 -> K base (l>=16?8:0),
// elems 8..15 -> K base+16. Two aligned 16B LDS reads.
__device__ inline v16bf frag_a(const __bf16* lds, int m0, int lane) {
  int row = m0 + (lane & 15);
  int kh  = (lane >> 4) << 3;
  const __bf16* p = lds + row * LDK + kh;
  ABu u;
  u.h[0] = *reinterpret_cast<const v8bf*>(p);
  u.h[1] = *reinterpret_cast<const v8bf*>(p + 16);
  return u.v;
}
// B fragment: 32x16 (KxN) from LDS tile stored [N][K]. lane l col = l&15;
// elems 0..7 -> K base (l>=16?8:0), elems 8..15 -> K base+16 (matches frag_a).
__device__ inline v16bf frag_b(const __bf16* lds, int n0, int lane) {
  int row = n0 + (lane & 15);
  int k0  = (lane >> 4) << 3;
  const __bf16* p = lds + row * LDK + k0;
  ABu u;
  u.h[0] = *reinterpret_cast<const v8bf*>(p);
  u.h[1] = *reinterpret_cast<const v8bf*>(p + 16);
  return u.v;
}

// ---------------------------------------------------------------------------
// Weight pre-pack: fp32 [Cout][K] -> bf16 [Mp][Kp], zero padded.
// ---------------------------------------------------------------------------
__global__ void k_packw(const float* __restrict__ W, __bf16* __restrict__ Wp,
                        int Cout, int K, int Kp, int Mp) {
  long i = blockIdx.x * 256L + threadIdx.x;
  long tot = (long)Mp * Kp;
  if (i >= tot) return;
  int m = (int)(i / Kp), k = (int)(i - (long)m * Kp);
  float v = (m < Cout && k < K) ? W[(long)m * K + k] : 0.f;
  Wp[i] = (__bf16)v;
}

// ---------------------------------------------------------------------------
// Implicit-GEMM convolution with WMMA, double-buffered LDS pipeline.
//   M = Cout (padded Mp), N = B*Ho*Wo, K = Cin*ks*ks (Kp = ceil32)
// Block: 128 threads / 4 waves; tile 64(M) x 128(N); wave computes 32x64.
// ---------------------------------------------------------------------------
__global__ __launch_bounds__(128) void k_conv_wmma(
    const float* __restrict__ X, const __bf16* __restrict__ Wp,
    const float* __restrict__ Bias, const float* __restrict__ Res,
    float* __restrict__ Out,
    int Cin, int H, int Win, int Cout, int Ho, int Wo,
    int ks, int stride, int pad, int K, int Kp)
{
  __shared__ __align__(16) __bf16 Al[2][TM * LDK];
  __shared__ __align__(16) __bf16 Bl[2][TN * LDK];

  const int tid  = threadIdx.x;
  const int lane = tid & 31;
  const int wid  = tid >> 5;
  const int wm   = (wid >> 1) * 32;
  const int wn   = (wid & 1) * 64;
  const int n0   = blockIdx.x * TN;
  const int m0   = blockIdx.y * TM;
  const int Npix = Ho * Wo;

  // K-invariant im2col decode for this thread's B row (one row per thread)
  const int gn  = n0 + tid;
  const int bB  = gn / Npix;
  const int pix = gn - bB * Npix;
  const int oy  = pix / Wo;
  const int ox  = pix - oy * Wo;
  const int iyb = oy * stride - pad;
  const int ixb = ox * stride - pad;
  const float* Xb = X + (long)bB * Cin * H * Win;

  // incremental k -> (ci,ky,kx) state, advances monotonically over all tiles
  int ci = 0, ky = 0, kx = 0, krun = 0;

  v8f acc[2][4];
  for (int i = 0; i < 2; ++i)
    for (int j = 0; j < 4; ++j)
      for (int r = 0; r < 8; ++r) acc[i][j][r] = 0.f;

  auto loadA = [&](int buf, int kt) {
    int gm   = tid >> 1;
    int half = (tid & 1) << 4;           // 0 or 16 elements
    const __bf16* src = Wp + (long)(m0 + gm) * Kp + kt * TK + half;
    __bf16* dst = &Al[buf][gm * LDK + half];
    copy16_to_lds(src,     dst);         // 8 bf16 (16B)
    copy16_to_lds(src + 8, dst + 8);     // 8 bf16 (16B)
  };

  auto loadB = [&](int buf) {
    __bf16* row = &Bl[buf][tid * LDK];
    for (int ch = 0; ch < 4; ++ch) {
      P8u pk;
      for (int j = 0; j < 8; ++j) {
        float v = 0.f;
        if (krun < K) {
          int iy = iyb + ky, ix = ixb + kx;
          if (iy >= 0 && iy < H && ix >= 0 && ix < Win)
            v = Xb[((long)ci * H + iy) * Win + ix];
        }
        pk.e[j] = (__bf16)v;
        if (ks == 3) { if (++kx == 3) { kx = 0; if (++ky == 3) { ky = 0; ++ci; } } }
        else ++ci;
        ++krun;
      }
      *reinterpret_cast<v8bf*>(row + ch * 8) = pk.v;
    }
  };

  auto compute = [&](int buf) {
    const __bf16* Ab = Al[buf];
    const __bf16* Bb = Bl[buf];
    v16bf a0 = frag_a(Ab, wm,      lane);
    v16bf a1 = frag_a(Ab, wm + 16, lane);
    v16bf b0 = frag_b(Bb, wn,      lane);
    v16bf b1 = frag_b(Bb, wn + 16, lane);
    v16bf b2 = frag_b(Bb, wn + 32, lane);
    v16bf b3 = frag_b(Bb, wn + 48, lane);
    acc[0][0] = wmma_bf16(a0, b0, acc[0][0]);
    acc[0][1] = wmma_bf16(a0, b1, acc[0][1]);
    acc[0][2] = wmma_bf16(a0, b2, acc[0][2]);
    acc[0][3] = wmma_bf16(a0, b3, acc[0][3]);
    acc[1][0] = wmma_bf16(a1, b0, acc[1][0]);
    acc[1][1] = wmma_bf16(a1, b1, acc[1][1]);
    acc[1][2] = wmma_bf16(a1, b2, acc[1][2]);
    acc[1][3] = wmma_bf16(a1, b3, acc[1][3]);
  };

  const int nt = Kp / TK;
  loadA(0, 0);
  loadB(0);
  wait_async_lds();
  __syncthreads();
  for (int t = 0; t < nt; ++t) {
    int cur = t & 1;
    if (t + 1 < nt) { loadA(1 - cur, t + 1); loadB(1 - cur); }
    compute(cur);
    wait_async_lds();
    __syncthreads();
  }

  // ---- epilogue: bias (+residual), scatter to NCHW
  for (int i = 0; i < 2; ++i)
    for (int j = 0; j < 4; ++j) {
      int mbase = m0 + wm + i * 16 + ((lane >> 4) << 3);
      int nn    = n0 + wn + j * 16 + (lane & 15);
      int b = nn / Npix, p2 = nn - b * Npix;
      for (int r = 0; r < 8; ++r) {
        int m = mbase + r;
        if (m < Cout) {
          float v = acc[i][j][r] + Bias[m];
          long idx = ((long)b * Cout + m) * Npix + p2;
          if (Res) v += Res[idx];
          Out[idx] = v;
        }
      }
    }
}

// ---------------------------------------------------------------------------
// GroupNorm (32 groups) + optional SiLU. One block per (batch, group).
// ---------------------------------------------------------------------------
__global__ __launch_bounds__(256) void k_gn(
    const float* __restrict__ X, const float* __restrict__ Gw,
    const float* __restrict__ Bw, float* __restrict__ Out,
    int C, int HW, int do_silu)
{
  int cg = C >> 5;
  int b = blockIdx.x >> 5;
  int g = blockIdx.x & 31;
  long base = ((long)b * C + (long)g * cg) * HW;
  int cnt = cg * HW;
  __shared__ float s1[256], s2[256];
  float a = 0.f, q = 0.f;
  for (int i = threadIdx.x; i < cnt; i += 256) {
    float v = X[base + i]; a += v; q += v * v;
  }
  s1[threadIdx.x] = a; s2[threadIdx.x] = q; __syncthreads();
  for (int o = 128; o >= 1; o >>= 1) {
    if (threadIdx.x < o) { s1[threadIdx.x] += s1[threadIdx.x + o]; s2[threadIdx.x] += s2[threadIdx.x + o]; }
    __syncthreads();
  }
  float mean = s1[0] / cnt;
  float var  = s2[0] / cnt - mean * mean;
  float inv  = rsqrtf(var + 1e-5f);
  for (int i = threadIdx.x; i < cnt; i += 256) {
    int c = g * cg + i / HW;
    float v = (X[base + i] - mean) * inv * Gw[c] + Bw[c];
    if (do_silu) v = v / (1.f + expf(-v));
    Out[base + i] = v;
  }
}

// ---------------------------------------------------------------------------
// Self-attention (nonlocal core): B=256 blocks, C=256, HW=64.
// ---------------------------------------------------------------------------
__global__ __launch_bounds__(256) void k_attn(
    const float* __restrict__ Q, const float* __restrict__ Kq,
    const float* __restrict__ V, float* __restrict__ Out)
{
  __shared__ float S[64][65];
  __shared__ float T0[64][65];
  __shared__ float T1[64][65];
  int b = blockIdx.x, t = threadIdx.x;
  const long bb = (long)b * 256 * 64;
  float part[16];
  for (int p = 0; p < 16; ++p) part[p] = 0.f;

  for (int c0 = 0; c0 < 256; c0 += 64) {
    for (int i = t; i < 4096; i += 256) { int cc = i >> 6, n = i & 63; T0[cc][n] = Q[bb + (long)(c0 + cc) * 64 + n]; }
    for (int i = t; i < 4096; i += 256) { int cc = i >> 6, m = i & 63; T1[cc][m] = Kq[bb + (long)(c0 + cc) * 64 + m]; }
    __syncthreads();
    for (int p = 0; p < 16; ++p) {
      int idx = t * 16 + p, n = idx >> 6, m = idx & 63;
      float s = 0.f;
      for (int cc = 0; cc < 64; ++cc) s += T0[cc][n] * T1[cc][m];
      part[p] += s;
    }
    __syncthreads();
  }
  for (int p = 0; p < 16; ++p) { int idx = t * 16 + p; S[idx >> 6][idx & 63] = part[p] * (1.0f / 16.0f); }
  __syncthreads();
  if (t < 64) {
    float mx = -1e30f;
    for (int m = 0; m < 64; ++m) mx = fmaxf(mx, S[t][m]);
    float sum = 0.f;
    for (int m = 0; m < 64; ++m) { float e = expf(S[t][m] - mx); S[t][m] = e; sum += e; }
    float inv = 1.f / sum;
    for (int m = 0; m < 64; ++m) S[t][m] *= inv;
  }
  __syncthreads();
  for (int c0 = 0; c0 < 256; c0 += 64) {
    for (int i = t; i < 4096; i += 256) { int cc = i >> 6, m = i & 63; T0[cc][m] = V[bb + (long)(c0 + cc) * 64 + m]; }
    __syncthreads();
    for (int p = 0; p < 16; ++p) {
      int idx = t * 16 + p, cc = idx >> 6, n = idx & 63;
      float o = 0.f;
      for (int m = 0; m < 64; ++m) o += T0[cc][m] * S[n][m];
      Out[bb + (long)(c0 + cc) * 64 + n] = o;
    }
    __syncthreads();
  }
}

// ---------------------------------------------------------------------------
// VQ: one block per latent vector row. 128 codes, 64 dims.
// ---------------------------------------------------------------------------
__global__ __launch_bounds__(128) void k_quant(
    const float* __restrict__ Fi, const float* __restrict__ Tab,
    float* __restrict__ Zq, int* __restrict__ IdxOut, float* __restrict__ accSlot, int s)
{
  __shared__ float zf[64];
  __shared__ float dmin[128];
  __shared__ int   didx[128];
  int r = blockIdx.x, ss = s * s;
  int b = r / ss, pix = r - b * ss;
  int t = threadIdx.x;
  if (t < 64) zf[t] = Fi[((long)b * 64 + t) * ss + pix];
  __syncthreads();
  float d = 0.f;
  const float* tr = Tab + t * 64;
  for (int c = 0; c < 64; ++c) { float df = zf[c] - tr[c]; d += df * df; }
  dmin[t] = d; didx[t] = t; __syncthreads();
  for (int o = 64; o >= 1; o >>= 1) {
    if (t < o) {
      float ov = dmin[t + o]; int oi = didx[t + o];
      if (ov < dmin[t] || (ov == dmin[t] && oi < didx[t])) { dmin[t] = ov; didx[t] = oi; }
    }
    __syncthreads();
  }
  int best = didx[0];
  if (t < 64) Zq[((long)b * 64 + t) * ss + pix] = Tab[best * 64 + t];
  if (t == 0) {
    IdxOut[r] = best;
    float e = 0.f;
    const float* tb = Tab + best * 64;
    for (int c = 0; c < 64; ++c) { float df = zf[c] - tb[c]; e += df * df; }
    atomicAdd(accSlot, e);
  }
}

// ---------------------------------------------------------------------------
// Bilinear resize (torch align_corners=False).
// ---------------------------------------------------------------------------
__global__ void k_resize(const float* __restrict__ In, float* __restrict__ Out,
                         int C, int ih, int iw, int oh, int ow)
{
  long i = blockIdx.x * 256L + threadIdx.x;
  long tot = 256L * C * oh * ow;
  if (i >= tot) return;
  int ox = i % ow; long r = i / ow;
  int oy = r % oh; r /= oh;
  int c = r % C; int b = (int)(r / C);
  float sy = (oy + 0.5f) * (float)ih / oh - 0.5f; if (sy < 0.f) sy = 0.f;
  int y0 = (int)floorf(sy); if (y0 > ih - 1) y0 = ih - 1;
  int y1 = y0 + 1; if (y1 > ih - 1) y1 = ih - 1;
  float ly = sy - y0;
  float sx = (ox + 0.5f) * (float)iw / ow - 0.5f; if (sx < 0.f) sx = 0.f;
  int x0 = (int)floorf(sx); if (x0 > iw - 1) x0 = iw - 1;
  int x1 = x0 + 1; if (x1 > iw - 1) x1 = iw - 1;
  float lx = sx - x0;
  const float* p = In + ((long)b * C + c) * ih * iw;
  float v = (1.f - ly) * ((1.f - lx) * p[y0 * iw + x0] + lx * p[y0 * iw + x1]) +
            ly         * ((1.f - lx) * p[y1 * iw + x0] + lx * p[y1 * iw + x1]);
  Out[i] = v;
}

// ---------------------------------------------------------------------------
// Elementwise helpers
// ---------------------------------------------------------------------------
__global__ void k_zero(float* p, long n) {
  long i = blockIdx.x * 256L + threadIdx.x; if (i < n) p[i] = 0.f;
}
__global__ void k_copy(const float* a, float* o, long n) {
  long i = blockIdx.x * 256L + threadIdx.x; if (i < n) o[i] = a[i];
}
__global__ void k_fupdate(float* f, float* fh, const float* co, long n) {
  long i = blockIdx.x * 256L + threadIdx.x;
  if (i < n) { f[i] -= co[i]; fh[i] += co[i]; }
}
__global__ void k_up2(const float* __restrict__ In, float* __restrict__ Out,
                      int C, int H, int W) {
  long i = blockIdx.x * 256L + threadIdx.x;
  long tot = 256L * C * 4 * H * W;
  if (i >= tot) return;
  int ow = 2 * W, oh = 2 * H;
  int x = i % ow; long r = i / ow;
  int y = r % oh; r /= oh;
  int c = r % C; int b = (int)(r / C);
  Out[i] = In[(((long)b * C + c) * H + (y >> 1)) * W + (x >> 1)];
}
__global__ __launch_bounds__(256) void k_mse(const float* A, const float* B, float* slot, long n) {
  __shared__ float sm[256];
  long i0 = blockIdx.x * 256L + threadIdx.x;
  float s = 0.f;
  for (long i = i0; i < n; i += (long)gridDim.x * 256L) { float d = A[i] - B[i]; s += d * d; }
  sm[threadIdx.x] = s; __syncthreads();
  for (int o = 128; o >= 1; o >>= 1) {
    if (threadIdx.x < o) sm[threadIdx.x] += sm[threadIdx.x + o];
    __syncthreads();
  }
  if (threadIdx.x == 0) atomicAdd(slot, sm[0]);
}
__global__ void k_final(const float* acc, float* out) {
  if (threadIdx.x == 0 && blockIdx.x == 0) {
    out[0] = 1.25f * (acc[0] / 16384.f + acc[1] / 65536.f +
                      acc[2] / 262144.f + acc[3] / 1048576.f);  // q_loss
    out[1] = acc[4] / 1048576.f;                                 // f_loss
    out[2] = acc[5] / 262144.f;                                  // r_loss
  }
}

// ---------------------------------------------------------------------------
// Host orchestration
// ---------------------------------------------------------------------------
struct ConvP { const float* w; const float* b; };
struct GnP   { const float* g; const float* b; };
struct ResP  { ConvP c1, c2; GnP g1, g2; };
struct NlP   { GnP gn; ConvP q, k, v, o; };

extern "C" void kernel_launch(void* const* d_in, const int* in_sizes, int n_in,
                              void* d_out, int out_size, void* d_ws, size_t ws_size,
                              hipStream_t stream) {
  (void)in_sizes; (void)n_in; (void)out_size; (void)ws_size;
  auto F = [&](int i) { return (const float*)d_in[i]; };
  auto mkconv = [&](int bi) { ConvP c; c.b = F(bi); c.w = F(bi + 1); return c; };
  auto mkgn   = [&](int bi) { GnP g; g.b = F(bi); g.g = F(bi + 1); return g; };
  auto mkres  = [&](int bi) { ResP r; r.c1 = mkconv(bi); r.c2 = mkconv(bi + 2);
                              r.g1 = mkgn(bi + 4); r.g2 = mkgn(bi + 6); return r; };
  auto mknl   = [&](int bi) { NlP n; n.gn = mkgn(bi); n.k = mkconv(bi + 2);
                              n.o = mkconv(bi + 4); n.q = mkconv(bi + 6);
                              n.v = mkconv(bi + 8); return n; };

  // JAX pytree (sorted-key) flat order: x, convs[0..3]{b,w}, dec{...}, enc{...}, table
  const float* x = F(0);
  ConvP scale_convs[4] = { mkconv(1), mkconv(3), mkconv(5), mkconv(7) };
  ConvP dec_conv1 = mkconv(9),  dec_conv2 = mkconv(11);
  GnP   dec_gn    = mkgn(13);
  NlP   dec_nl    = mknl(15);
  ResP  dec_res1    = mkres(25), dec_res2    = mkres(33);
  ResP  dec_res_nl1 = mkres(41), dec_res_nl2 = mkres(49);
  ConvP dec_up1 = mkconv(57), dec_up2 = mkconv(59);
  ConvP enc_down1 = mkconv(61), enc_down2 = mkconv(63);
  GnP   enc_gn  = mkgn(65);
  ConvP enc_init = mkconv(67), enc_latent = mkconv(69);
  NlP   enc_nl = mknl(71);
  ResP  enc_res1 = mkres(81), enc_res2 = mkres(89);
  ResP  enc_rend1 = mkres(97), enc_rend2 = mkres(105);
  const float* table = F(113);

  // Workspace carve-up (floats)
  float* ws = (float*)d_ws;
  size_t off = 0;
  float* G0 = ws + off; off += 33554432;   // up to 128ch @ 32x32
  float* G1 = ws + off; off += 16777216;
  float* G2 = ws + off; off += 16777216;
  float* Qb = ws + off; off += 4194304;
  float* Kb = ws + off; off += 4194304;
  float* Vb = ws + off; off += 4194304;
  float* Forig  = ws + off; off += 1048576;
  float* Ff     = ws + off; off += 1048576;
  float* Fhat   = ws + off; off += 1048576;
  float* Ffi    = ws + off; off += 1048576;
  float* Fzq    = ws + off; off += 1048576;
  float* Fzfull = ws + off; off += 1048576;
  float* Fco    = ws + off; off += 1048576;
  float* Acc    = ws + off; off += 16;
  __bf16* Wpk   = (__bf16*)(ws + off); off += 294912;  // 256x2304 bf16 max

  float* xhat  = (float*)d_out;                 // [256,1,32,32]
  float* louts = (float*)d_out + 262144;        // q,f,r
  int*   idxb  = (int*)d_out;
  const int idx_off[4] = { 262147, 262403, 263427, 267523 };
  const int scales[4]  = { 1, 2, 4, 8 };

  auto ew = [&](long n) { return dim3((unsigned)((n + 255) / 256)); };

  auto conv = [&](const float* in, ConvP p, float* out, const float* res,
                  int Cin, int H, int W, int Cout, int ks, int stride, int pad) {
    int Ho = (H + 2 * pad - ks) / stride + 1;
    int Wo = (W + 2 * pad - ks) / stride + 1;
    int K  = Cin * ks * ks;
    int Kp = (K + 31) & ~31;
    int Mp = (Cout + 63) & ~63;
    long N = 256L * Ho * Wo;
    k_packw<<<ew((long)Mp * Kp), 256, 0, stream>>>(p.w, Wpk, Cout, K, Kp, Mp);
    dim3 grid((unsigned)(N / TN), (unsigned)(Mp / TM));
    k_conv_wmma<<<grid, dim3(128), 0, stream>>>(in, Wpk, p.b, res, out,
        Cin, H, W, Cout, Ho, Wo, ks, stride, pad, K, Kp);
  };
  auto gn = [&](const float* in, GnP p, float* out, int C, int HW, int silu) {
    k_gn<<<dim3(256 * 32), dim3(256), 0, stream>>>(in, p.g, p.b, out, C, HW, silu);
  };
  // result lands in `t`
  auto resb = [&](const float* in, float* t, float* s, ResP rp, int C, int H, int W) {
    gn(in, rp.g1, s, C, H * W, 1);
    conv(s, rp.c1, t, nullptr, C, H, W, C, 3, 1, 1);
    gn(t, rp.g2, s, C, H * W, 1);
    conv(s, rp.c2, t, in, C, H, W, C, 3, 1, 1);
  };
  auto nlblock = [&](const float* in /*kept*/, NlP p, float* y, float* attn_out, float* out) {
    gn(in, p.gn, y, 256, 64, 0);
    conv(y, p.q, Qb, nullptr, 256, 8, 8, 256, 1, 1, 0);
    conv(y, p.k, Kb, nullptr, 256, 8, 8, 256, 1, 1, 0);
    conv(y, p.v, Vb, nullptr, 256, 8, 8, 256, 1, 1, 0);
    k_attn<<<dim3(256), dim3(256), 0, stream>>>(Qb, Kb, Vb, attn_out);
    conv(attn_out, p.o, out, in, 256, 8, 8, 256, 1, 1, 0);
  };

  // ---------------- Encoder ----------------
  conv(x, enc_init, G1, nullptr, 1, 32, 32, 64, 3, 1, 1);          // G1 64@32
  resb(G1, G2, G0, enc_res1, 64, 32, 32);                          // G2
  conv(G2, enc_down1, G1, nullptr, 64, 32, 32, 128, 3, 2, 1);      // G1 128@16
  resb(G1, G2, G0, enc_res2, 128, 16, 16);                         // G2
  conv(G2, enc_down2, G1, nullptr, 128, 16, 16, 256, 3, 2, 1);     // G1 256@8
  resb(G1, G2, G0, enc_rend1, 256, 8, 8);                          // G2
  nlblock(G2, enc_nl, G0, G1, G0);                                 // G0
  resb(G0, G2, G1, enc_rend2, 256, 8, 8);                          // G2
  gn(G2, enc_gn, G1, 256, 64, 1);
  conv(G1, enc_latent, Forig, nullptr, 256, 8, 8, 64, 3, 1, 1);    // f_orig

  k_copy<<<ew(1048576), 256, 0, stream>>>(Forig, Ff, 1048576);
  k_zero<<<ew(1048576), 256, 0, stream>>>(Fhat, 1048576);
  k_zero<<<1, 32, 0, stream>>>(Acc, 16);

  // ---------------- Multi-scale VQ ----------------
  for (int sid = 0; sid < 4; ++sid) {
    int s = scales[sid];
    long nfi = 256L * 64 * s * s;
    k_resize<<<ew(nfi), 256, 0, stream>>>(Ff, Ffi, 64, 8, 8, s, s);
    k_quant<<<dim3((unsigned)(256 * s * s)), dim3(128), 0, stream>>>(
        Ffi, table, Fzq, idxb + idx_off[sid], Acc + sid, s);
    k_resize<<<ew(1048576), 256, 0, stream>>>(Fzq, Fzfull, 64, s, s, 8, 8);
    conv(Fzfull, scale_convs[sid], Fco, nullptr, 64, 8, 8, 64, 3, 1, 1);
    k_fupdate<<<ew(1048576), 256, 0, stream>>>(Ff, Fhat, Fco, 1048576);
  }

  // ---------------- Decoder ----------------
  conv(Fhat, dec_conv1, G1, nullptr, 64, 8, 8, 256, 3, 1, 1);      // G1 256@8
  resb(G1, G2, G0, dec_res_nl1, 256, 8, 8);                        // G2
  nlblock(G2, dec_nl, G0, G1, G0);                                 // G0
  resb(G0, G2, G1, dec_res_nl2, 256, 8, 8);                        // G2
  resb(G2, G1, G0, dec_res1, 256, 8, 8);                           // G1
  k_up2<<<ew(16777216), 256, 0, stream>>>(G1, G2, 256, 8, 8);      // G2 256@16
  conv(G2, dec_up1, G1, nullptr, 256, 16, 16, 128, 3, 1, 1);       // G1 128@16
  resb(G1, G2, G0, dec_res2, 128, 16, 16);                         // G2
  k_up2<<<ew(33554432), 256, 0, stream>>>(G2, G0, 128, 16, 16);    // G0 128@32
  conv(G0, dec_up2, G1, nullptr, 128, 32, 32, 64, 3, 1, 1);        // G1 64@32
  gn(G1, dec_gn, G2, 64, 1024, 1);
  conv(G2, dec_conv2, xhat, nullptr, 64, 32, 32, 1, 3, 1, 1);      // x_hat -> d_out

  // ---------------- Losses ----------------
  k_mse<<<dim3(2048), dim3(256), 0, stream>>>(Fhat, Forig, Acc + 4, 1048576);
  k_mse<<<dim3(2048), dim3(256), 0, stream>>>(xhat, x, Acc + 5, 262144);
  k_final<<<1, 32, 0, stream>>>(Acc, louts);
}
